// VQGanAttention_10411000725840
// MI455X (gfx1250) — compile-verified, matrix operand order checked by reading, not run on hardware
//
#include <hip/hip_runtime.h>
#include <hip/hip_bf16.h>

// ---------------------------------------------------------------------------
// VQGAN attention block for gfx1250 (MI455X), all matmuls via
// v_wmma_f32_16x16x32_f16 (wave32 WMMA). CPB bias MLP collapsed to the exact
// 63x63 relative-offset table. Flash-attention so sim [8,8,1024,1024] is never
// materialized. GEMMs register-blocked 16x64 per wave (A-fragment reuse x4),
// fragment loads batched ahead of WMMA groups for load/compute overlap.
// ---------------------------------------------------------------------------

typedef __attribute__((ext_vector_type(16))) _Float16 v16h;
typedef __attribute__((ext_vector_type(8)))  float    v8f;

#define WMMA_F16(a, b, c) \
  __builtin_amdgcn_wmma_f32_16x16x32_f16(false, (a), false, (b), (short)0, (c), false, false)

union H16 { v16h v; unsigned int u[8]; };

// Load a 16x32 f16 fragment per the CDNA5 A/B VGPR layout:
// lane l: row/col = l&15, K-half = l>>4; K pairs contiguous -> 2x b128 loads.
// Source is "row-major with contiguous K": element (m,k) at p[m*ld + k].
__device__ __forceinline__ v16h load_frag(const _Float16* __restrict__ p,
                                          int row0, int ld, int k0, int lane) {
  int m  = lane & 15;
  int kh = (lane >> 4) & 1;
  const _Float16* base = p + (size_t)(row0 + m) * (size_t)ld + (size_t)(k0 + (kh << 3));
  H16 f;
#pragma unroll
  for (int pp = 0; pp < 8; ++pp) {
    int k = ((pp & 3) << 1) + ((pp >> 2) << 4);  // K pattern {0,2,4,6,16,18,20,22}
    f.u[pp] = *(const unsigned int*)(base + k);
  }
  return f.v;
}

__device__ __forceinline__ v8f zero8() {
  v8f z = {0.f, 0.f, 0.f, 0.f, 0.f, 0.f, 0.f, 0.f};
  return z;
}

// ---------------------------------------------------------------------------
// 0) f32 -> f16 convert (weights)
// ---------------------------------------------------------------------------
__global__ void cvt_f16_kernel(const float* __restrict__ in, _Float16* __restrict__ out, int n) {
  int i = blockIdx.x * 256 + threadIdx.x;
  if (i < n) out[i] = (_Float16)in[i];
}

// ---------------------------------------------------------------------------
// 1) Channel LayerNorm -> xnT[b][pix][c] (f16), block = 32 pixels x 8 c-rakes
// ---------------------------------------------------------------------------
__global__ void ln_kernel(const float* __restrict__ x, const float* __restrict__ gamma,
                          _Float16* __restrict__ xnT) {
  int blk  = blockIdx.x;            // 8*32 blocks
  int b    = blk >> 5;
  int pix0 = (blk & 31) << 5;
  int tx   = threadIdx.x & 31;      // pixel lane (coalesced)
  int ty   = threadIdx.x >> 5;      // channel rake 0..7
  int pix  = pix0 + tx;

  const float* xb = x + ((size_t)b << 19) + pix;   // [b][c][pix], c stride 1024
  float s = 0.f, s2 = 0.f;
  for (int c = ty; c < 512; c += 8) {
    float v = xb[(size_t)c << 10];
    s += v; s2 += v * v;
  }
  __shared__ float rs[8][32], rs2[8][32];
  __shared__ float mean_s[32], rstd_s[32];
  rs[ty][tx] = s; rs2[ty][tx] = s2;
  __syncthreads();
  if (ty == 0) {
    float a = 0.f, a2 = 0.f;
#pragma unroll
    for (int t = 0; t < 8; ++t) { a += rs[t][tx]; a2 += rs2[t][tx]; }
    float mean = a * (1.f / 512.f);
    float var  = a2 * (1.f / 512.f) - mean * mean;   // biased variance (jnp.var)
    mean_s[tx] = mean;
    rstd_s[tx] = rsqrtf(var + 1e-5f);
  }
  __syncthreads();
  float mean = mean_s[tx], rstd = rstd_s[tx];
  _Float16* op = xnT + ((size_t)b << 19) + (size_t)pix * 512;
  for (int c = ty; c < 512; c += 8) {
    float v = xb[(size_t)c << 10];
    op[c] = (_Float16)((v - mean) * rstd * gamma[c]);
  }
}

// ---------------------------------------------------------------------------
// 2) CPB bias: exact 63x63 table per head (bias depends only on (dy,dx))
// ---------------------------------------------------------------------------
__global__ void cpb_kernel(const float* __restrict__ w0, const float* __restrict__ b0,
                           const float* __restrict__ w1, const float* __restrict__ b1,
                           const float* __restrict__ w2, const float* __restrict__ b2,
                           float* __restrict__ biasTab) {
  __shared__ float h0s[64][129];
  int idx = blockIdx.x * 64 + threadIdx.x;
  if (idx >= 3969) return;
  int di = idx / 63 - 31;
  int dj = idx % 63 - 31;
  float r0 = ((di > 0) ? 1.f : (di < 0) ? -1.f : 0.f) * __logf(fabsf((float)di) + 1.f);
  float r1 = ((dj > 0) ? 1.f : (dj < 0) ? -1.f : 0.f) * __logf(fabsf((float)dj) + 1.f);
  float* h0 = h0s[threadIdx.x];
  for (int d = 0; d < 128; ++d) {
    float a = w0[d * 2] * r0 + w0[d * 2 + 1] * r1 + b0[d];
    h0[d] = (a >= 0.f) ? a : 0.1f * a;               // leaky 0.1
  }
  float oh[8];
#pragma unroll
  for (int h = 0; h < 8; ++h) oh[h] = b2[h];
  for (int d = 0; d < 128; ++d) {
    float a = b1[d];
    const float* wr = w1 + d * 128;
    for (int c = 0; c < 128; ++c) a += wr[c] * h0[c];
    a = (a >= 0.f) ? a : 0.1f * a;
#pragma unroll
    for (int h = 0; h < 8; ++h) oh[h] += w2[h * 128 + d] * a;
  }
#pragma unroll
  for (int h = 0; h < 8; ++h) biasTab[h * 3969 + idx] = oh[h];
}

// ---------------------------------------------------------------------------
// 3) QKV GEMM: qkv[o,pix] = sum_c Wqkv[o,c] * xnT[pix,c]
//    wave = 16(m) x 64(n) strip; per k-step: batch af + 4 bf loads, 4 WMMAs.
//    writes qT/kT[bh][pix][d] (q pre-scaled by 1/8) and v[bh][d][pix], f16.
// ---------------------------------------------------------------------------
__global__ void qkv_gemm_kernel(const _Float16* __restrict__ W, const _Float16* __restrict__ xnT,
                                _Float16* __restrict__ qT, _Float16* __restrict__ kT,
                                _Float16* __restrict__ vv) {
  int lane = threadIdx.x & 31;
  int wv   = threadIdx.x >> 5;                  // 4 waves
  int m0   = blockIdx.x << 4;                   // output row tile (0..1535)
  int n0   = (blockIdx.y << 8) + (wv << 6);     // 64-wide pixel strip
  int b    = blockIdx.z;
  const _Float16* Bm = xnT + ((size_t)b << 19);
  v8f acc[4];
#pragma unroll
  for (int nt = 0; nt < 4; ++nt) acc[nt] = zero8();
#pragma unroll 2
  for (int k0 = 0; k0 < 512; k0 += 32) {
    v16h af = load_frag(W, m0, 512, k0, lane);  // reused across 4 n-tiles
    v16h bf[4];
#pragma unroll
    for (int nt = 0; nt < 4; ++nt) bf[nt] = load_frag(Bm, n0 + (nt << 4), 512, k0, lane);
#pragma unroll
    for (int nt = 0; nt < 4; ++nt) acc[nt] = WMMA_F16(af, bf[nt], acc[nt]);
  }
  int which = m0 >> 9;                // 0=q 1=k 2=v (tile never straddles)
  int h     = (m0 >> 6) & 7;
  int dbase = m0 & 63;
  int hi    = lane >> 4;
  size_t bh = (size_t)(b * 8 + h);
#pragma unroll
  for (int nt = 0; nt < 4; ++nt) {
    int pix = n0 + (nt << 4) + (lane & 15);
#pragma unroll
    for (int r = 0; r < 8; ++r) {
      int d = dbase + r + (hi << 3);
      float vval = acc[nt][r];
      if (which == 0)      qT[(bh * 1024 + pix) * 64 + d] = (_Float16)(vval * 0.125f);
      else if (which == 1) kT[(bh * 1024 + pix) * 64 + d] = (_Float16)vval;
      else                 vv[(bh * 64 + d) * 1024 + pix] = (_Float16)vval;
    }
  }
}

// ---------------------------------------------------------------------------
// 4) Flash attention: block = (b, h, 64-query tile), 4 waves x 16 queries
// ---------------------------------------------------------------------------
__global__ void attn_kernel(const _Float16* __restrict__ qT, const _Float16* __restrict__ kT,
                            const _Float16* __restrict__ vv, const float* __restrict__ biasTab,
                            _Float16* __restrict__ aoT) {
  __shared__ float biasS[3969];
  __shared__ __align__(4) _Float16 pS[4][16][72];  // per-wave P tile (C->A transpose)

  int lane = threadIdx.x & 31;
  int wv   = threadIdx.x >> 5;
  int h    = blockIdx.y;
  int b    = blockIdx.z;
  for (int t = threadIdx.x; t < 3969; t += 128) biasS[t] = biasTab[h * 3969 + t];
  __syncthreads();

  size_t bh = (size_t)(b * 8 + h);
  const _Float16* qb = qT + bh * 1024 * 64;
  const _Float16* kb = kT + bh * 1024 * 64;
  const _Float16* vb = vv + bh * 64 * 1024;
  int i0 = (blockIdx.x << 6) + (wv << 4);

  v16h qa0 = load_frag(qb, i0, 64, 0,  lane);
  v16h qa1 = load_frag(qb, i0, 64, 32, lane);

  v8f o[4];
  float mrow[8], lrow[8];
#pragma unroll
  for (int dt = 0; dt < 4; ++dt) o[dt] = zero8();
#pragma unroll
  for (int r = 0; r < 8; ++r) { mrow[r] = -3.0e38f; lrow[r] = 0.f; }

  int hi = lane >> 4, nl = lane & 15;

  for (int jt = 0; jt < 16; ++jt) {
    int j0 = jt << 6;
    v8f s[4];
    // S = Q^T K: batch fragment loads 2 n-tiles at a time, then 4 WMMAs
#pragma unroll
    for (int np = 0; np < 2; ++np) {
      v16h kf[4];
#pragma unroll
      for (int u = 0; u < 2; ++u) {
        int nt = (np << 1) + u;
        kf[u * 2 + 0] = load_frag(kb, j0 + (nt << 4), 64, 0,  lane);
        kf[u * 2 + 1] = load_frag(kb, j0 + (nt << 4), 64, 32, lane);
      }
#pragma unroll
      for (int u = 0; u < 2; ++u) {
        int nt = (np << 1) + u;
        s[nt] = zero8();
        s[nt] = WMMA_F16(qa0, kf[u * 2 + 0], s[nt]);
        s[nt] = WMMA_F16(qa1, kf[u * 2 + 1], s[nt]);
      }
    }
    // bias add (exact table gather from LDS)
#pragma unroll
    for (int nt = 0; nt < 4; ++nt) {
      int j = j0 + (nt << 4) + nl;
      int yj = j >> 5, xj = j & 31;
#pragma unroll
      for (int r = 0; r < 8; ++r) {
        int i = i0 + r + (hi << 3);
        s[nt][r] += biasS[((i >> 5) - yj + 31) * 63 + ((i & 31) - xj + 31)];
      }
    }
    // online softmax: row max over 4 frags + 16-lane half reduction
    float nm[8];
#pragma unroll
    for (int r = 0; r < 8; ++r)
      nm[r] = fmaxf(fmaxf(s[0][r], s[1][r]), fmaxf(s[2][r], s[3][r]));
#pragma unroll
    for (int off = 8; off >= 1; off >>= 1)
#pragma unroll
      for (int r = 0; r < 8; ++r) nm[r] = fmaxf(nm[r], __shfl_xor(nm[r], off, 32));
    float fac[8];
#pragma unroll
    for (int r = 0; r < 8; ++r) {
      float M = fmaxf(mrow[r], nm[r]);
      fac[r] = __expf(mrow[r] - M);
      mrow[r] = M;
      lrow[r] *= fac[r];
    }
#pragma unroll
    for (int dt = 0; dt < 4; ++dt)
#pragma unroll
      for (int r = 0; r < 8; ++r) o[dt][r] *= fac[r];

    float rs[8];
#pragma unroll
    for (int r = 0; r < 8; ++r) rs[r] = 0.f;
#pragma unroll
    for (int nt = 0; nt < 4; ++nt)
#pragma unroll
      for (int r = 0; r < 8; ++r) {
        float p = __expf(s[nt][r] - mrow[r]);
        rs[r] += p;
        pS[wv][r + (hi << 3)][(nt << 4) + nl] = (_Float16)p;
      }
#pragma unroll
    for (int off = 8; off >= 1; off >>= 1)
#pragma unroll
      for (int r = 0; r < 8; ++r) rs[r] += __shfl_xor(rs[r], off, 32);
#pragma unroll
    for (int r = 0; r < 8; ++r) lrow[r] += rs[r];

    // P @ V^T via LDS transpose (wave-private region, dscnt-ordered)
    v16h pa0 = load_frag(&pS[wv][0][0], 0, 72, 0,  lane);
    v16h pa1 = load_frag(&pS[wv][0][0], 0, 72, 32, lane);
#pragma unroll
    for (int dp = 0; dp < 2; ++dp) {
      v16h vf[4];
#pragma unroll
      for (int u = 0; u < 2; ++u) {
        int dt = (dp << 1) + u;
        vf[u * 2 + 0] = load_frag(vb, dt << 4, 1024, j0,      lane);
        vf[u * 2 + 1] = load_frag(vb, dt << 4, 1024, j0 + 32, lane);
      }
#pragma unroll
      for (int u = 0; u < 2; ++u) {
        int dt = (dp << 1) + u;
        o[dt] = WMMA_F16(pa0, vf[u * 2 + 0], o[dt]);
        o[dt] = WMMA_F16(pa1, vf[u * 2 + 1], o[dt]);
      }
    }
  }
  // epilogue: normalize and write aoT[b][pix][h*64+d] (f16, GEMM-B layout)
#pragma unroll
  for (int dt = 0; dt < 4; ++dt)
#pragma unroll
    for (int r = 0; r < 8; ++r) {
      int i = i0 + r + (hi << 3);
      int d = (dt << 4) + nl;
      aoT[((size_t)b << 19) + (size_t)i * 512 + h * 64 + d] = (_Float16)(o[dt][r] / lrow[r]);
    }
}

// ---------------------------------------------------------------------------
// 5) Output GEMM + residual: out[b,c,pix] = sum_i Wout[c,i]*aoT[pix,i] + x
//    wave = 16(m) x 64(n) strip; batched fragment loads, A reuse x4.
// ---------------------------------------------------------------------------
__global__ void out_gemm_kernel(const _Float16* __restrict__ W, const _Float16* __restrict__ aoT,
                                const float* __restrict__ x, float* __restrict__ out) {
  int lane = threadIdx.x & 31;
  int wv   = threadIdx.x >> 5;
  int m0   = blockIdx.x << 4;                   // channel tile (0..511)
  int n0   = (blockIdx.y << 8) + (wv << 6);     // 64-wide pixel strip
  int b    = blockIdx.z;
  const _Float16* Bm = aoT + ((size_t)b << 19);
  v8f acc[4];
#pragma unroll
  for (int nt = 0; nt < 4; ++nt) acc[nt] = zero8();
#pragma unroll 2
  for (int k0 = 0; k0 < 512; k0 += 32) {
    v16h af = load_frag(W, m0, 512, k0, lane);
    v16h bf[4];
#pragma unroll
    for (int nt = 0; nt < 4; ++nt) bf[nt] = load_frag(Bm, n0 + (nt << 4), 512, k0, lane);
#pragma unroll
    for (int nt = 0; nt < 4; ++nt) acc[nt] = WMMA_F16(af, bf[nt], acc[nt]);
  }
  int hi = lane >> 4;
#pragma unroll
  for (int nt = 0; nt < 4; ++nt) {
    int pix = n0 + (nt << 4) + (lane & 15);
#pragma unroll
    for (int r = 0; r < 8; ++r) {
      int c = m0 + r + (hi << 3);
      size_t off = ((size_t)(b * 512 + c) << 10) + pix;
      out[off] = acc[nt][r] + x[off];
    }
  }
}

// ---------------------------------------------------------------------------
extern "C" void kernel_launch(void* const* d_in, const int* in_sizes, int n_in,
                              void* d_out, int out_size, void* d_ws, size_t ws_size,
                              hipStream_t stream) {
  const float* x     = (const float*)d_in[0];
  const float* gamma = (const float*)d_in[1];
  const float* w_qkv = (const float*)d_in[2];
  const float* w_out = (const float*)d_in[3];
  const float* cw0   = (const float*)d_in[4];
  const float* cb0   = (const float*)d_in[5];
  const float* cw1   = (const float*)d_in[6];
  const float* cb1   = (const float*)d_in[7];
  const float* cw2   = (const float*)d_in[8];
  const float* cb2   = (const float*)d_in[9];
  float* out = (float*)d_out;

  char* ws = (char*)d_ws;
  size_t off = 0;
  auto alloc = [&](size_t bytes) -> void* {
    void* p = ws + off;
    off += (bytes + 255) & ~(size_t)255;
    return p;
  };
  const size_t HALF = 2;
  _Float16* xnT   = (_Float16*)alloc((size_t)8 * 1024 * 512 * HALF);   // 8 MB
  _Float16* qT    = (_Float16*)alloc((size_t)8 * 8 * 1024 * 64 * HALF);
  _Float16* kT    = (_Float16*)alloc((size_t)8 * 8 * 1024 * 64 * HALF);
  _Float16* vv    = (_Float16*)alloc((size_t)8 * 8 * 64 * 1024 * HALF);
  _Float16* aoT   = (_Float16*)alloc((size_t)8 * 1024 * 512 * HALF);
  _Float16* wqkvh = (_Float16*)alloc((size_t)1536 * 512 * HALF);
  _Float16* wouth = (_Float16*)alloc((size_t)512 * 512 * HALF);
  float*    biasT = (float*)alloc((size_t)8 * 3969 * sizeof(float));
  (void)in_sizes; (void)n_in; (void)out_size; (void)ws_size;

  cvt_f16_kernel<<<(1536 * 512 + 255) / 256, 256, 0, stream>>>(w_qkv, wqkvh, 1536 * 512);
  cvt_f16_kernel<<<(512 * 512 + 255) / 256, 256, 0, stream>>>(w_out, wouth, 512 * 512);
  ln_kernel<<<8 * 32, 256, 0, stream>>>(x, gamma, xnT);
  cpb_kernel<<<63, 64, 0, stream>>>(cw0, cb0, cw1, cb1, cw2, cb2, biasT);
  qkv_gemm_kernel<<<dim3(96, 4, 8), 128, 0, stream>>>(wqkvh, xnT, qT, kT, vv);
  attn_kernel<<<dim3(16, 8, 8), 128, 0, stream>>>(qT, kT, vv, biasT, aoT);
  out_gemm_kernel<<<dim3(32, 4, 8), 128, 0, stream>>>(wouth, aoT, x, out);
}
